// Head_58574763983796
// MI455X (gfx1250) — matrix operand order repaired and checked by reading, not compiled, not run on hardware
//
#include <hip/hip_runtime.h>
#include <hip/hip_bf16.h>

// Single-head causal attention for MI455X (gfx1250), wave32 + WMMA bf16.
// B=2048, T=64, C=384, H=64. One workgroup per batch element, 8 waves.

typedef __attribute__((ext_vector_type(16))) __bf16 v16bf;
typedef __attribute__((ext_vector_type(8)))  float  v8f;

#define TT 64
#define HH 64
#define CC 384
#define CHUNK 128
#define XS_LD 136      // bf16 elems per row (128 + pad)
#define QK_LD 72       // bf16 elems per row (64 + pad)
#define SS_LD 65       // f32 elems per row (64 + pad)
#define PS_LD 72       // bf16 elems per row

#define SMEM_XS_OFF 0
#define SMEM_Q_OFF  17408                 // 64*136*2
#define SMEM_K_OFF  (SMEM_Q_OFF + 64*QK_LD*2)
#define SMEM_V_OFF  (SMEM_K_OFF + 64*QK_LD*2)
#define SMEM_S_OFF  (SMEM_V_OFF + 64*QK_LD*2)   // 45056
#define SMEM_BYTES  (SMEM_S_OFF + 64*SS_LD*4)   // 61696 < 64KB

__device__ __forceinline__ v8f wmma_bf16(v16bf a, v16bf b, v8f c) {
    return __builtin_amdgcn_wmma_f32_16x16x32_bf16(
        /*neg_a=*/false, a, /*neg_b=*/false, b,
        /*c_mod=*/(short)0, c, /*reuse_a=*/false, /*reuse_b=*/false);
}

// A-matrix 16x32 bf16 fragment: lane holds M = lane&15,
// K = e + 16*(lane>=16).  `row` points at start of M-row, k0 = kk + 16*hi.
__device__ __forceinline__ v16bf frag_a(const __bf16* row, int k0) {
    v16bf f;
#pragma unroll
    for (int e = 0; e < 16; ++e) f[e] = row[k0 + e];
    return f;
}

// B-matrix 32x16 bf16 fragment: lane holds N = lane&15, K = e + 16*(lane>=16).
// `row` points at 16 contiguous K values for this lane's N, k0 = kk + 16*hi.
__device__ __forceinline__ v16bf frag_b_rowK(const __bf16* row, int k0) {
    v16bf f;
#pragma unroll
    for (int e = 0; e < 16; ++e) f[e] = row[k0 + e];
    return f;
}

__global__ void cvt_weights(const float* __restrict__ wq,
                            const float* __restrict__ wk,
                            const float* __restrict__ wv,
                            __bf16* __restrict__ o) {
    const int n = HH * CC;
    int i = blockIdx.x * blockDim.x + threadIdx.x;
    if (i < n) {
        o[i]         = (__bf16)wq[i];
        o[n + i]     = (__bf16)wk[i];
        o[2 * n + i] = (__bf16)wv[i];
    }
}

__launch_bounds__(256)
__global__ void attn_head(const float* __restrict__ x,
                          const __bf16* __restrict__ wbf,  // [3][64][384] bf16 (Q,K,V)
                          float* __restrict__ out) {
    __shared__ __align__(16) char smem[SMEM_BYTES];
    __bf16* Xs = (__bf16*)(smem + SMEM_XS_OFF);
    __bf16* Ps = (__bf16*)(smem + SMEM_XS_OFF);   // overlays Xs (dead after proj)
    __bf16* Qs = (__bf16*)(smem + SMEM_Q_OFF);
    __bf16* Ks = (__bf16*)(smem + SMEM_K_OFF);
    __bf16* Vs = (__bf16*)(smem + SMEM_V_OFF);
    float*  Ss = (float*)(smem + SMEM_S_OFF);

    const int b    = blockIdx.x;
    const int tid  = threadIdx.x;
    const int lane = tid & 31;
    const int wave = tid >> 5;
    const int lr   = lane & 15;        // row (A) or col (B/C) within tile
    const int hi   = (lane >> 4) & 1;  // half-wave select

    const float SCALE = 3.26598632f;   // 64 * 384^-0.5

    // ---------------- Phase B: QKV projections (X[64x384] * W^T[384x64]) ----
    v8f acc[6];
#pragma unroll
    for (int j = 0; j < 6; ++j) acc[j] = (v8f){};

    const size_t xbase = (size_t)b * TT * CC;

#pragma unroll 1
    for (int ch = 0; ch < CC / CHUNK; ++ch) {
        __syncthreads();
        // load + convert a 64x128 fp32 chunk of x into bf16 LDS
#pragma unroll
        for (int i = 0; i < 8; ++i) {
            int idx = tid + 256 * i;          // 0..2047 -> 64 rows x 32 float4
            int t   = idx >> 5;
            int c4  = idx & 31;
            const float4 v = *(const float4*)(x + xbase + (size_t)t * CC + ch * CHUNK + c4 * 4);
            __bf16* dst = Xs + t * XS_LD + c4 * 4;
            dst[0] = (__bf16)v.x; dst[1] = (__bf16)v.y;
            dst[2] = (__bf16)v.z; dst[3] = (__bf16)v.w;
        }
        __syncthreads();

#pragma unroll
        for (int j = 0; j < 6; ++j) {
            int job  = wave + 8 * j;          // 48 jobs: 3 mats x 16 tiles
            int mat  = job >> 4;
            int tile = job & 15;
            int tm   = tile >> 2, tn = tile & 3;
            const __bf16* arow = Xs + (tm * 16 + lr) * XS_LD;
            const __bf16* brow = wbf + ((mat * HH + tn * 16 + lr) * CC) + ch * CHUNK;
#pragma unroll
            for (int kk = 0; kk < CHUNK; kk += 32) {
                v16bf a = frag_a(arow, kk + hi * 16);
                v16bf bb = frag_b_rowK(brow, kk + hi * 16);
                acc[j] = wmma_bf16(a, bb, acc[j]);
            }
        }
    }

    // write Q/K/V tiles (bf16) to LDS; C layout: M = r + 8*hi, N = lr
#pragma unroll
    for (int j = 0; j < 6; ++j) {
        int job  = wave + 8 * j;
        int mat  = job >> 4;
        int tile = job & 15;
        int tm   = tile >> 2, tn = tile & 3;
        __bf16* dst = (mat == 0) ? Qs : (mat == 1) ? Ks : Vs;
#pragma unroll
        for (int r = 0; r < 8; ++r)
            dst[(tm * 16 + hi * 8 + r) * QK_LD + tn * 16 + lr] = (__bf16)acc[j][r];
    }
    __syncthreads();

    // ---------------- Phase C: S = Q * K^T * scale (causal tiles only) ------
#pragma unroll
    for (int j = 0; j < 2; ++j) {
        int tile = wave + 8 * j;
        int tm = tile >> 2, tn = tile & 3;
        if (tn > tm) continue;                // fully-masked tile
        v8f c = (v8f){};
        const __bf16* arow = Qs + (tm * 16 + lr) * QK_LD;
        const __bf16* brow = Ks + (tn * 16 + lr) * QK_LD;  // B[k][n]=K[n][k]
#pragma unroll
        for (int kk = 0; kk < HH; kk += 32) {
            v16bf a  = frag_a(arow, kk + hi * 16);
            v16bf bb = frag_b_rowK(brow, kk + hi * 16);
            c = wmma_bf16(a, bb, c);
        }
#pragma unroll
        for (int r = 0; r < 8; ++r)
            Ss[(tm * 16 + hi * 8 + r) * SS_LD + tn * 16 + lr] = c[r] * SCALE;
    }
    __syncthreads();

    // ---------------- Phase D: row softmax with causal mask -----------------
    if (tid < TT) {
        int t = tid;
        float mx = -3.0e38f;
        for (int s = 0; s <= t; ++s) mx = fmaxf(mx, Ss[t * SS_LD + s]);
        float sum = 0.f;
        for (int s = 0; s <= t; ++s) {
            float e = __expf(Ss[t * SS_LD + s] - mx);
            Ss[t * SS_LD + s] = e;
            sum += e;
        }
        float inv = 1.0f / sum;
        for (int s = 0; s < TT; ++s)
            Ps[t * PS_LD + s] = (s <= t) ? (__bf16)(Ss[t * SS_LD + s] * inv)
                                         : (__bf16)0.0f;
    }
    __syncthreads();

    // ---------------- Phase E: O = P * V, store fp32 to global --------------
#pragma unroll
    for (int j = 0; j < 2; ++j) {
        int tile = wave + 8 * j;
        int tm = tile >> 2, tn = tile & 3;
        v8f c = (v8f){};
        const __bf16* arow = Ps + (tm * 16 + lr) * PS_LD;
#pragma unroll
        for (int kk = 0; kk < TT; kk += 32) {
            v16bf a = frag_a(arow, kk + hi * 16);
            v16bf bb;                          // B[k][n] = V[s=k][h=n] (column walk)
            int kb = kk + hi * 16;
#pragma unroll
            for (int e = 0; e < 16; ++e)
                bb[e] = Vs[(kb + e) * QK_LD + tn * 16 + lr];
            c = wmma_bf16(a, bb, c);
        }
#pragma unroll
        for (int r = 0; r < 8; ++r)
            out[((size_t)b * TT + tm * 16 + hi * 8 + r) * HH + tn * 16 + lr] = c[r];
    }
}

extern "C" void kernel_launch(void* const* d_in, const int* in_sizes, int n_in,
                              void* d_out, int out_size, void* d_ws, size_t ws_size,
                              hipStream_t stream) {
    const float* x  = (const float*)d_in[0];
    const float* Wk = (const float*)d_in[1];
    const float* Wq = (const float*)d_in[2];
    const float* Wv = (const float*)d_in[3];
    float* out = (float*)d_out;
    __bf16* wbf = (__bf16*)d_ws;   // [3][64][384] bf16 in order Q,K,V

    const int nW = HH * CC;        // 24576 per matrix
    cvt_weights<<<(nW + 255) / 256, 256, 0, stream>>>(Wq, Wk, Wv, wbf);

    const int B = in_sizes[0] / (TT * CC);   // 2048
    attn_head<<<B, 256, 0, stream>>>(x, wbf, out);
}
